// MTPool_55808805044922
// MI455X (gfx1250) — matrix operand order repaired and checked by reading, not compiled, not run on hardware
//
#include <hip/hip_runtime.h>
#include <math.h>

typedef __attribute__((ext_vector_type(16))) _Float16 v16h;
typedef __attribute__((ext_vector_type(8)))  _Float16 v8h;
typedef __attribute__((ext_vector_type(8)))  float    v8f;

#define B_TOT 8192
#define NN    29
#define FF    147
#define DD    128
#define DCAT  429
#define DCATP 448
#define BN_EPS 1e-5f

union AFrag { v16h v; v8h h[2]; };
union CFrag { v8f  v; float f[8]; };

__device__ __forceinline__ float wave_sum(float v){
  #pragma unroll
  for (int off = 16; off > 0; off >>= 1) v += __shfl_xor(v, off, 32);
  return v;
}
__device__ __forceinline__ float wave_max(float v){
  #pragma unroll
  for (int off = 16; off > 0; off >>= 1) v = fmaxf(v, __shfl_xor(v, off, 32));
  return v;
}
__device__ __forceinline__ v8f wmma_f16(v16h a, v16h b, v8f c){
  return __builtin_amdgcn_wmma_f32_16x16x32_f16(false, a, false, b, (short)0, c, false, false);
}

// ---------------------------------------------------------------------------
// Kernel 0: prep — build f16 K-major transposes of Wg (429x128 -> 128x448,
// zero-padded K) and Wm1 (128x128 -> 128x128), zero the stat accumulators.
// ---------------------------------------------------------------------------
__global__ void prep_kernel(const float* __restrict__ Wg,
                            const float* __restrict__ Wm1,
                            _Float16* __restrict__ WgT,
                            _Float16* __restrict__ Wm1T,
                            float* __restrict__ statz){
  int idx = blockIdx.x * blockDim.x + threadIdx.x;
  if (idx < DD * DCATP){
    int c = idx / DCATP, k = idx - c * DCATP;
    WgT[idx] = (k < DCAT) ? (_Float16)Wg[(size_t)k * DD + c] : (_Float16)0.0f;
  }
  if (idx < DD * DD){
    int c = idx / DD, k = idx - c * DD;
    Wm1T[idx] = (_Float16)Wm1[(size_t)k * DD + c];
  }
  if (idx < 372) statz[idx] = 0.0f;  // s1(58) + s2(58) + s3(256)
}

// ---------------------------------------------------------------------------
// Kernel 1: fused conv branches (k=3,5,7 along F) + PReLU -> x1 (f16),
// plus per-n sum / sumsq for batch_norm over (B, d_cat).
// One block per b; wave w handles rows n = w, w+8, ...
// ---------------------------------------------------------------------------
__global__ void conv_kernel(const float* __restrict__ X,
    const float* w1, const float* b1, const float* w2, const float* b2,
    const float* w3, const float* b3, const float* a_cnn,
    _Float16* __restrict__ x1, float* s1sum, float* s1sq){
  __shared__ float sX[NN * FF];
  int b = blockIdx.x, tid = threadIdx.x;
  for (int i = tid; i < NN * FF; i += 256) sX[i] = X[(size_t)b * NN * FF + i];
  float c1[3], c2[5], c3[7];
  #pragma unroll
  for (int t = 0; t < 3; t++) c1[t] = w1[t];
  #pragma unroll
  for (int t = 0; t < 5; t++) c2[t] = w2[t];
  #pragma unroll
  for (int t = 0; t < 7; t++) c3[t] = w3[t];
  float bb1 = b1[0], bb2 = b2[0], bb3 = b3[0], al = a_cnn[0];
  __syncthreads();
  int wave = tid >> 5, lane = tid & 31;
  for (int n = wave; n < NN; n += 8){
    const float* row = sX + n * FF;
    float s = 0.f, s2 = 0.f;
    for (int j = lane; j < DCAT; j += 32){
      float v;
      if (j < 145){
        v = bb1;
        #pragma unroll
        for (int t = 0; t < 3; t++) v += row[j + t] * c1[t];
      } else if (j < 288){
        int pq = j - 145; v = bb2;
        #pragma unroll
        for (int t = 0; t < 5; t++) v += row[pq + t] * c2[t];
      } else {
        int pq = j - 288; v = bb3;
        #pragma unroll
        for (int t = 0; t < 7; t++) v += row[pq + t] * c3[t];
      }
      v = (v >= 0.f) ? v : al * v;
      x1[((size_t)b * NN + n) * DCAT + j] = (_Float16)v;
      s += v; s2 += v * v;
    }
    s = wave_sum(s); s2 = wave_sum(s2);
    if (lane == 0){ atomicAdd(&s1sum[n], s); atomicAdd(&s1sq[n], s2); }
  }
}

// ---------------------------------------------------------------------------
// Kernel 2: finalize stats1 + fold the whole attention-head stack into one
// vector u[d] = sum_h head_w[h] * centroids[h,0,d] / (||centroids[h]|| + 1e-8)
// ---------------------------------------------------------------------------
__global__ void finalize1_kernel(const float* s1sum, const float* s1sq,
    float* mean1, float* inv1,
    const float* __restrict__ cent, const float* head_w, float* uvec){
  __shared__ float nrm[4];
  int tid = threadIdx.x, wave = tid >> 5, lane = tid & 31;
  if (wave < 4){
    float p = 0.f;
    for (int d = lane; d < DD; d += 32){ float c = cent[wave * DD + d]; p += c * c; }
    p = wave_sum(p);
    if (lane == 0) nrm[wave] = sqrtf(p) + 1e-8f;
  }
  __syncthreads();
  if (tid < NN){
    const float cnt = (float)B_TOT * (float)DCAT;
    float m = s1sum[tid] / cnt;
    float v = s1sq[tid] / cnt - m * m;
    mean1[tid] = m; inv1[tid] = rsqrtf(v + BN_EPS);
  }
  if (tid < DD){
    float u = 0.f;
    #pragma unroll
    for (int h = 0; h < 4; h++) u += head_w[h] * cent[h * DD + tid] / nrm[h];
    uvec[tid] = u;
  }
}

// ---------------------------------------------------------------------------
// Kernel 3: the big one. Per b:
//   T = batchnorm1(x1[b]) @ Wg           (29x429 @ 429x128, f16 WMMA, K padded 448)
//   G = prelu(corr[b] @ T + bg)          (29x29 @ 29x128, VALU)
// plus per-n sum/sumsq of G for batch_norm2.
// 8 waves; wave w owns output column tile w (16 cols), both row tiles (0-15,16-31).
// A fragments from LDS (documented 16-bit A layout), B fragments from K-major WgT.
// ---------------------------------------------------------------------------
__global__ void gemm_kernel(const _Float16* __restrict__ x1,
    const float* mean1, const float* inv1,
    const _Float16* __restrict__ WgT, const float* __restrict__ corr,
    const float* bg, const float* a_gnn,
    float* __restrict__ Gbuf, float* s2sum, float* s2sq){
  __shared__ _Float16 sA[32 * DCATP];   // 28672 B: normalized x1 rows (padded)
  __shared__ float    sT[32 * DD];      // 16384 B: T = xnorm @ Wg
  __shared__ float    sC[NN * NN];      //  3364 B: corr[b]
  __shared__ float    sS[2 * NN];
  int b = blockIdx.x, tid = threadIdx.x;
  for (int i = tid; i < 32 * DCATP; i += 256){
    int m = i / DCATP, k = i - m * DCATP;
    float v = 0.f;
    if (m < NN && k < DCAT)
      v = ((float)x1[((size_t)b * NN + m) * DCAT + k] - mean1[m]) * inv1[m];
    sA[i] = (_Float16)v;
  }
  for (int i = tid; i < NN * NN; i += 256) sC[i] = corr[(size_t)b * NN * NN + i];
  if (tid < 2 * NN) sS[tid] = 0.f;
  __syncthreads();

  int wave = tid >> 5, lane = tid & 31;
  int laneM = lane & 15, hi = (lane >> 4) & 1;
  CFrag acc0, acc1;
  #pragma unroll
  for (int i = 0; i < 8; i++){ acc0.f[i] = 0.f; acc1.f[i] = 0.f; }
  const _Float16* aBase0 = sA + laneM * DCATP + hi * 8;
  const _Float16* aBase1 = sA + (16 + laneM) * DCATP + hi * 8;
  const _Float16* bBase  = WgT + (size_t)(wave * 16 + laneM) * DCATP + hi * 8;
  for (int kc = 0; kc < DCATP / 32; kc++){
    int kb = kc * 32;
    AFrag a0, a1;
    a0.h[0] = *(const v8h*)(aBase0 + kb);
    a0.h[1] = *(const v8h*)(aBase0 + kb + 16);
    a1.h[0] = *(const v8h*)(aBase1 + kb);
    a1.h[1] = *(const v8h*)(aBase1 + kb + 16);
    AFrag bq; bq.h[0] = *(const v8h*)(bBase + kb); bq.h[1] = *(const v8h*)(bBase + kb + 16);
    v16h bf = bq.v;
    acc0.v = wmma_f16(a0.v, bf, acc0.v);
    acc1.v = wmma_f16(a1.v, bf, acc1.v);
  }
  int Ncol = wave * 16 + laneM;
  int Mb = hi * 8;
  #pragma unroll
  for (int v = 0; v < 8; v++){
    sT[(Mb + v) * DD + Ncol]      = acc0.f[v];
    sT[(16 + Mb + v) * DD + Ncol] = acc1.f[v];
  }
  __syncthreads();

  float ag = a_gnn[0];
  for (int idx = tid; idx < NN * DD; idx += 256){
    int n = idx >> 7, d = idx & 127;
    float acc = bg[d];
    #pragma unroll 1
    for (int m = 0; m < NN; m++) acc += sC[n * NN + m] * sT[m * DD + d];
    float g = (acc >= 0.f) ? acc : ag * acc;
    Gbuf[((size_t)b * NN + n) * DD + d] = g;
    atomicAdd(&sS[n], g);
    atomicAdd(&sS[NN + n], g * g);
  }
  __syncthreads();
  if (tid < NN){ atomicAdd(&s2sum[tid], sS[tid]); atomicAdd(&s2sq[tid], sS[NN + tid]); }
}

__global__ void finalize2_kernel(const float* s, const float* q, float* mean2, float* inv2){
  int t = threadIdx.x;
  if (t < NN){
    const float cnt = (float)B_TOT * (float)DD;
    float m = s[t] / cnt, v = q[t] / cnt - m * m;
    mean2[t] = m; inv2[t] = rsqrtf(v + BN_EPS);
  }
}

// ---------------------------------------------------------------------------
// Kernel 5: per b: batch_norm2, attention logits a_n = <u, x_n>/(||x_n||+1e-8),
// softmax over n, pooled = sum_n C_n x_n, proj = pooled @ Wp + bp,
// plus per-d sum/sumsq for batch_norm3.
// ---------------------------------------------------------------------------
__global__ void pool_kernel(const float* __restrict__ Gbuf,
    const float* mean2, const float* inv2, const float* __restrict__ uvec,
    const float* __restrict__ Wp, const float* bp,
    float* __restrict__ P, float* s3sum, float* s3sq){
  __shared__ float sx[NN * DD];
  __shared__ float sa[32];
  __shared__ float spool[DD];
  int b = blockIdx.x, tid = threadIdx.x;
  for (int i = tid; i < NN * DD; i += 128){
    int n = i >> 7;
    sx[i] = (Gbuf[(size_t)b * NN * DD + i] - mean2[n]) * inv2[n];
  }
  __syncthreads();
  int wave = tid >> 5, lane = tid & 31;
  for (int n = wave; n < NN; n += 4){
    float nn = 0.f, dp = 0.f;
    for (int d = lane; d < DD; d += 32){
      float xv = sx[n * DD + d];
      nn += xv * xv; dp += uvec[d] * xv;
    }
    nn = wave_sum(nn); dp = wave_sum(dp);
    if (lane == 0) sa[n] = dp / (sqrtf(nn) + 1e-8f);
  }
  __syncthreads();
  if (wave == 0){
    float v = (lane < NN) ? sa[lane] : -3.4e38f;
    float mx = wave_max(v);
    float e = (lane < NN) ? __expf(v - mx) : 0.f;
    float s = wave_sum(e);
    if (lane < NN) sa[lane] = e / s;
  }
  __syncthreads();
  if (tid < DD){
    float pd = 0.f;
    #pragma unroll 1
    for (int n = 0; n < NN; n++) pd += sa[n] * sx[n * DD + tid];
    spool[tid] = pd;
  }
  __syncthreads();
  if (tid < DD){
    float acc = bp[tid];
    #pragma unroll 1
    for (int k = 0; k < DD; k++) acc += spool[k] * Wp[(size_t)k * DD + tid];
    P[(size_t)b * DD + tid] = acc;
    atomicAdd(&s3sum[tid], acc);
    atomicAdd(&s3sq[tid], acc * acc);
  }
}

__global__ void finalize3_kernel(const float* s, const float* q, float* mean3, float* inv3){
  int t = threadIdx.x;
  if (t < DD){
    const float cnt = (float)B_TOT;
    float m = s[t] / cnt, v = q[t] / cnt - m * m;
    mean3[t] = m; inv3[t] = rsqrtf(v + BN_EPS);
  }
}

// ---------------------------------------------------------------------------
// Kernel 7: MLP head, 16 batch rows per block tile via WMMA:
//   h = prelu(batchnorm3(P) @ Wm1 + bm1);  y = sigmoid(h @ Wm2 + bm2)
// 8 waves; wave w owns h columns 16w..16w+15; row-dot with Wm2 reduced via
// LDS float atomics.
// ---------------------------------------------------------------------------
__global__ void mlp_kernel(const float* __restrict__ P,
    const float* mean3, const float* inv3,
    const _Float16* __restrict__ Wm1T, const float* bm1,
    const float* a_mlp, const float* __restrict__ Wm2, const float* bm2,
    float* __restrict__ out){
  __shared__ _Float16 sZ[16 * DD];
  __shared__ float sH[16];
  int b0 = blockIdx.x * 16, tid = threadIdx.x;
  for (int i = tid; i < 16 * DD; i += 256){
    int r = i >> 7, k = i & 127;
    sZ[i] = (_Float16)((P[(size_t)(b0 + r) * DD + k] - mean3[k]) * inv3[k]);
  }
  if (tid < 16) sH[tid] = 0.f;
  __syncthreads();
  int wave = tid >> 5, lane = tid & 31;
  int laneM = lane & 15, hi = (lane >> 4) & 1;
  CFrag acc;
  #pragma unroll
  for (int i = 0; i < 8; i++) acc.f[i] = 0.f;
  const _Float16* aBase = sZ + laneM * DD + hi * 8;
  const _Float16* bBase = Wm1T + (size_t)(wave * 16 + laneM) * DD + hi * 8;
  #pragma unroll
  for (int kc = 0; kc < 4; kc++){
    int kb = kc * 32;
    AFrag a;
    a.h[0] = *(const v8h*)(aBase + kb);
    a.h[1] = *(const v8h*)(aBase + kb + 16);
    AFrag bq; bq.h[0] = *(const v8h*)(bBase + kb); bq.h[1] = *(const v8h*)(bBase + kb + 16);
    v16h bf = bq.v;
    acc.v = wmma_f16(a.v, bf, acc.v);
  }
  int Ncol = wave * 16 + laneM;
  float amlp = a_mlp[0], bc = bm1[Ncol], wc = Wm2[Ncol];
  int Mb = hi * 8;
  #pragma unroll
  for (int v = 0; v < 8; v++){
    float h = acc.f[v] + bc;
    h = (h >= 0.f) ? h : amlp * h;
    atomicAdd(&sH[Mb + v], h * wc);
  }
  __syncthreads();
  if (tid < 16){
    float y = 1.f / (1.f + __expf(-(sH[tid] + bm2[0])));
    out[b0 + tid] = y;
  }
}

// ---------------------------------------------------------------------------
extern "C" void kernel_launch(void* const* d_in, const int* in_sizes, int n_in,
                              void* d_out, int out_size, void* d_ws, size_t ws_size,
                              hipStream_t stream){
  (void)in_sizes; (void)n_in; (void)out_size; (void)ws_size;
  const float* X      = (const float*)d_in[0];
  const float* corr   = (const float*)d_in[1];
  const float* w_c1   = (const float*)d_in[2];
  const float* b_c1   = (const float*)d_in[3];
  const float* w_c2   = (const float*)d_in[4];
  const float* b_c2   = (const float*)d_in[5];
  const float* w_c3   = (const float*)d_in[6];
  const float* b_c3   = (const float*)d_in[7];
  const float* a_cnn  = (const float*)d_in[8];
  const float* Wg     = (const float*)d_in[9];
  const float* bg     = (const float*)d_in[10];
  const float* a_gnn  = (const float*)d_in[11];
  const float* cent   = (const float*)d_in[12];
  const float* head_w = (const float*)d_in[13];
  const float* Wp     = (const float*)d_in[14];
  const float* bp     = (const float*)d_in[15];
  const float* Wm1    = (const float*)d_in[16];
  const float* bm1    = (const float*)d_in[17];
  const float* a_mlp  = (const float*)d_in[18];
  const float* Wm2    = (const float*)d_in[19];
  const float* bm2    = (const float*)d_in[20];
  float* out = (float*)d_out;

  char* p = (char*)d_ws;
  auto take = [&](size_t bytes) -> char* {
    char* r = p; p += (bytes + 255) & ~(size_t)255; return r;
  };
  _Float16* x1    = (_Float16*)take((size_t)B_TOT * NN * DCAT * sizeof(_Float16)); // 204 MB
  float*    Gbuf  = (float*)   take((size_t)B_TOT * NN * DD   * sizeof(float));    // 122 MB
  float*    Pb    = (float*)   take((size_t)B_TOT * DD        * sizeof(float));    // 4.2 MB
  _Float16* WgT   = (_Float16*)take((size_t)DD * DCATP * sizeof(_Float16));        // 115 KB (L2-resident)
  _Float16* Wm1T  = (_Float16*)take((size_t)DD * DD    * sizeof(_Float16));
  float*    statz = (float*)   take(372 * sizeof(float));
  float*    aux   = (float*)   take(530 * sizeof(float));
  float *s1sum = statz,       *s1sq = statz + 29;
  float *s2sum = statz + 58,  *s2sq = statz + 87;
  float *s3sum = statz + 116, *s3sq = statz + 244;
  float *mean1 = aux,        *inv1 = aux + 29;
  float *mean2 = aux + 58,   *inv2 = aux + 87;
  float *mean3 = aux + 116,  *inv3 = aux + 244;
  float *uvec  = aux + 372;

  prep_kernel<<<224, 256, 0, stream>>>(Wg, Wm1, WgT, Wm1T, statz);
  conv_kernel<<<B_TOT, 256, 0, stream>>>(X, w_c1, b_c1, w_c2, b_c2, w_c3, b_c3,
                                         a_cnn, x1, s1sum, s1sq);
  finalize1_kernel<<<1, 128, 0, stream>>>(s1sum, s1sq, mean1, inv1, cent, head_w, uvec);
  gemm_kernel<<<B_TOT, 256, 0, stream>>>(x1, mean1, inv1, WgT, corr, bg, a_gnn,
                                         Gbuf, s2sum, s2sq);
  finalize2_kernel<<<1, 32, 0, stream>>>(s2sum, s2sq, mean2, inv2);
  pool_kernel<<<B_TOT, 128, 0, stream>>>(Gbuf, mean2, inv2, uvec, Wp, bp,
                                         Pb, s3sum, s3sq);
  finalize3_kernel<<<1, 128, 0, stream>>>(s3sum, s3sq, mean3, inv3);
  mlp_kernel<<<B_TOT / 16, 256, 0, stream>>>(Pb, mean3, inv3, Wm1T, bm1,
                                             a_mlp, Wm2, bm2, out);
}
